// Layer_74191265071820
// MI455X (gfx1250) — compile-verified, matrix operand order checked by reading, not run on hardware
//
#include <hip/hip_runtime.h>

// ---------------------------------------------------------------------------
// Problem constants (from reference)
// ---------------------------------------------------------------------------
constexpr int SEQ    = 512;   // L
constexpr int NBATCH = 2;     // N
constexpr int DMODEL = 256;
constexpr int NHEADS = 8;
constexpr int HDIM   = 32;
constexpr int ROWS   = SEQ * NBATCH;   // 1024 rows for projections (r = i*N + n)

typedef __attribute__((ext_vector_type(16))) __bf16 v16bf;
typedef __attribute__((ext_vector_type(8)))  __bf16 v8bf;
typedef __attribute__((ext_vector_type(8)))  float  v8f;
typedef __attribute__((ext_vector_type(4)))  float  v4f;

union BF16x16 { v16bf v; v8bf h[2]; __bf16 e[16]; };
union F32x4   { v4f v; float e[4]; };

// Load one WMMA 16-bit A/B operand (16x32 tile, row-major source):
// lane m (<16) holds row m, K = {k0..k0+7, k0+16..k0+23};
// lane m+16 holds row m, K = {k0+8..k0+15, k0+24..k0+31}.  (ISA 7.12.2)
static __device__ inline v16bf ld_tile(const __bf16* base, int stride,
                                       int row0, int col0, int lane) {
  int r = row0 + (lane & 15);
  int c = col0 + ((lane & 16) >> 1);      // +8 for upper half-wave
  const __bf16* p = base + (size_t)r * stride + c;
  BF16x16 u;
  u.h[0] = *(const v8bf*)(p);
  u.h[1] = *(const v8bf*)(p + 16);
  return u.v;
}

static __device__ inline v8f wmma_bf16(v16bf a, v16bf b, v8f c) {
  return __builtin_amdgcn_wmma_f32_16x16x32_bf16(false, a, false, b,
                                                 (short)0, c, false, false);
}

// ---------------------------------------------------------------------------
// K0: fp32 -> bf16 conversion (s and weight matrices)
// ---------------------------------------------------------------------------
__global__ void cvt_bf16_kernel(const float* __restrict__ src,
                                __bf16* __restrict__ dst, int n) {
  int i  = blockIdx.x * blockDim.x + threadIdx.x;
  int st = gridDim.x * blockDim.x;
  for (; i < n; i += st) dst[i] = (__bf16)src[i];
}

// Build zero-padded 16x256 bf16 copy of Wm (rows 8..15 = 0)
__global__ void prep_wm_kernel(const float* __restrict__ Wm,
                               __bf16* __restrict__ Wmb) {
  int i = blockIdx.x * blockDim.x + threadIdx.x;   // 0 .. 16*DMODEL-1
  if (i < 16 * DMODEL) {
    int h = i / DMODEL;
    Wmb[i] = (h < NHEADS) ? (__bf16)Wm[i] : (__bf16)0.0f;
  }
}

// ---------------------------------------------------------------------------
// K1: bias[n][h][i][j] = dot(rel_p[n][i][j][:], Wm[h][:])  via WMMA.
// The 536 MB streaming kernel.  One wave per 16-vector tile:
//   A = 16 rel_p vectors (fp32 -> bf16 on the fly, NT loads),
//   B = padded Wm^T (preloaded in registers), K = 256 = 8 x K32 WMMA steps.
// ~8 wave-instructions per KB streamed -> stays under the HBM roofline.
// ---------------------------------------------------------------------------
__global__ void bias_proj_wmma_kernel(const float* __restrict__ rel_p,
                                      const __bf16* __restrict__ Wmb,
                                      float* __restrict__ bias) {
  const int lane = threadIdx.x & 31;
  const int wid  = (blockIdx.x * blockDim.x + threadIdx.x) >> 5;
  const int nw   = (gridDim.x * blockDim.x) >> 5;

  // Preload all 8 B operands (Wm^T tiles): B[k][h] = Wm[h][k]
  v16bf bw[8];
  for (int kk = 0; kk < 8; ++kk)
    bw[kk] = ld_tile(Wmb, DMODEL, 0, kk * 32, lane);

  const int rowoff = (lane & 16) >> 1;
  const int col    = lane & 15;
  const int ntiles = NBATCH * SEQ * SEQ / 16;   // 32768

  for (int tile = wid; tile < ntiles; tile += nw) {
    const float* rp = rel_p + (size_t)tile * 16 * DMODEL;
    v8f acc = {0, 0, 0, 0, 0, 0, 0, 0};
    for (int kk = 0; kk < 8; ++kk) {
      int k0 = kk * 32;
      // lane covers row (lane&15), feats {k0+off..+7, k0+off+16..+23}, off=0|8
      const float* p = rp + (size_t)(lane & 15) * DMODEL + k0 + rowoff;
      F32x4 f0, f1, f2, f3;
      f0.v = __builtin_nontemporal_load((const v4f*)(p));
      f1.v = __builtin_nontemporal_load((const v4f*)(p + 4));
      f2.v = __builtin_nontemporal_load((const v4f*)(p + 16));
      f3.v = __builtin_nontemporal_load((const v4f*)(p + 20));
      BF16x16 a;
      for (int c = 0; c < 4; ++c) {
        a.e[c]      = (__bf16)f0.e[c];
        a.e[4 + c]  = (__bf16)f1.e[c];
        a.e[8 + c]  = (__bf16)f2.e[c];
        a.e[12 + c] = (__bf16)f3.e[c];
      }
      acc = wmma_bf16(a.v, bw[kk], acc);
    }
    // D: row m = local vector index, col = head (cols 8..15 are padding)
    if (col < NHEADS) {
      for (int g = 0; g < 8; ++g) {
        int r  = tile * 16 + g + rowoff;      // global (n,i,j) vector index
        int n  = r / (SEQ * SEQ);
        int ij = r % (SEQ * SEQ);
        bias[(size_t)(n * NHEADS + col) * SEQ * SEQ + ij] = acc[g];
      }
    }
  }
}

// ---------------------------------------------------------------------------
// K2: QKV projection via WMMA.  out[r][o] = sum_k s[r][k] * W[o][k]
// One wave per 16x16 output tile; 3072 wave-tasks total.
// Q -> qb[n][h][i][d], K -> kb[n][h][j][d], V -> vT[n][h][d][j] (transposed).
// ---------------------------------------------------------------------------
__global__ void qkv_gemm_kernel(const __bf16* __restrict__ sb,
                                const __bf16* __restrict__ Wqb,
                                const __bf16* __restrict__ Wkb,
                                const __bf16* __restrict__ Wvb,
                                __bf16* __restrict__ qb,
                                __bf16* __restrict__ kb,
                                __bf16* __restrict__ vT) {
  const int lane = threadIdx.x & 31;
  const int w    = blockIdx.x * (blockDim.x >> 5) + (threadIdx.x >> 5);
  const int m    = w >> 10;              // 0=Q, 1=K, 2=V
  const int rem  = w & 1023;
  const int r0   = (rem >> 4) << 4;
  const int o0   = (rem & 15) << 4;
  const __bf16* W = (m == 0) ? Wqb : (m == 1) ? Wkb : Wvb;

  v8f acc = {0, 0, 0, 0, 0, 0, 0, 0};
  for (int k0 = 0; k0 < DMODEL; k0 += 32) {
    v16bf a = ld_tile(sb, DMODEL, r0, k0, lane);   // A[m][k] = s[r0+m][k]
    v16bf b = ld_tile(W,  DMODEL, o0, k0, lane);   // B[k][n] = W[o0+n][k]
    acc = wmma_bf16(a, b, acc);
  }

  const int rowoff = (lane & 16) >> 1;
  const int col    = lane & 15;
  const int o      = o0 + col;
  const int h      = o >> 5;
  const int d      = o & 31;
  for (int g = 0; g < 8; ++g) {
    int r = r0 + g + rowoff;
    int i = r >> 1;          // r = i*NBATCH + n, NBATCH = 2
    int n = r & 1;
    __bf16 val = (__bf16)acc[g];
    if (m == 0)
      qb[(((size_t)(n * NHEADS + h) * SEQ + i) << 5) + d] = val;
    else if (m == 1)
      kb[(((size_t)(n * NHEADS + h) * SEQ + i) << 5) + d] = val;
    else
      vT[((size_t)(n * NHEADS + h) * HDIM + d) * SEQ + i] = val;
  }
}

// ---------------------------------------------------------------------------
// K3: fused attention per (i-tile, h, n).  4 waves / 128 threads.
// Phase 1: S = (Q Kt) * scale + bias  -> 16x512 fp32 tile in LDS (WMMA)
// Phase 2: row softmax (unnormalized exp -> bf16 LDS, keep row sums)
// Phase 3: O = P V via WMMA, normalize, write bf16 ctx
// ---------------------------------------------------------------------------
__global__ __launch_bounds__(128)
void attn_kernel(const __bf16* __restrict__ qb,
                 const __bf16* __restrict__ kb,
                 const __bf16* __restrict__ vT,
                 const float* __restrict__ bias,
                 __bf16* __restrict__ ctxb) {
  __shared__ __align__(16) float  sS[16 * SEQ];     // 32 KB
  __shared__ __align__(16) __bf16 pb[16 * SEQ];     // 16 KB
  __shared__ float rowsum[16];

  const int lane  = threadIdx.x & 31;
  const int wave  = threadIdx.x >> 5;
  const int itile = blockIdx.x;
  const int h     = blockIdx.y;
  const int n     = blockIdx.z;
  const int i0    = itile << 4;

  const __bf16* qh = qb + (((size_t)(n * NHEADS + h) * SEQ) << 5);
  const __bf16* kh = kb + (((size_t)(n * NHEADS + h) * SEQ) << 5);
  const __bf16* vh = vT + ((size_t)(n * NHEADS + h) * HDIM) * SEQ;
  const float*  bh = bias + ((size_t)(n * NHEADS + h) * SEQ + i0) * SEQ;

  const int rowoff = (lane & 16) >> 1;
  const int col    = lane & 15;
  const float scale = 0.0625f;          // 1/sqrt(DMODEL)

  // ---- Phase 1: scores ----
  v16bf aq = ld_tile(qh, HDIM, i0, 0, lane);
  for (int t = 0; t < 8; ++t) {
    int j0 = (wave + 4 * t) << 4;
    v16bf bk = ld_tile(kh, HDIM, j0, 0, lane);     // B[d][j] = k[j0+j][d]
    v8f acc = {0, 0, 0, 0, 0, 0, 0, 0};
    acc = wmma_bf16(aq, bk, acc);
    for (int g = 0; g < 8; ++g) {
      int row = g + rowoff;
      sS[row * SEQ + j0 + col] =
          acc[g] * scale + bh[(size_t)row * SEQ + j0 + col];
    }
  }
  __syncthreads();

  // ---- Phase 2: softmax (wave w owns rows 4w..4w+3) ----
  for (int q = 0; q < 4; ++q) {
    int row = wave * 4 + q;
    float mx = -3.4e38f;
    for (int j = lane; j < SEQ; j += 32) mx = fmaxf(mx, sS[row * SEQ + j]);
    for (int off = 16; off > 0; off >>= 1) mx = fmaxf(mx, __shfl_xor(mx, off, 32));
    float sum = 0.f;
    for (int j = lane; j < SEQ; j += 32) {
      float e = __expf(sS[row * SEQ + j] - mx);
      pb[row * SEQ + j] = (__bf16)e;
      sum += e;
    }
    for (int off = 16; off > 0; off >>= 1) sum += __shfl_xor(sum, off, 32);
    if (lane == 0) rowsum[row] = sum;
  }
  __syncthreads();

  // ---- Phase 3: O = P V (waves 0,1 each own a 16-wide d-tile) ----
  if (wave < 2) {
    int d0 = wave << 4;
    v8f acc = {0, 0, 0, 0, 0, 0, 0, 0};
    for (int c = 0; c < 16; ++c) {
      int j0 = c << 5;
      v16bf ap = ld_tile(pb, SEQ, 0, j0, lane);    // A[m][k] = P[m][j0+k]
      v16bf bv = ld_tile(vh, SEQ, d0, j0, lane);   // B[k][d] = vT[d0+d][j0+k]
      acc = wmma_bf16(ap, bv, acc);
    }
    for (int g = 0; g < 8; ++g) {
      int row = g + rowoff;
      float val = acc[g] * (1.0f / rowsum[row]);
      int i = i0 + row;
      ctxb[((size_t)(i * NBATCH + n) * DMODEL) + h * HDIM + d0 + col] =
          (__bf16)val;
    }
  }
}

// ---------------------------------------------------------------------------
// K4: output projection  out[r][o] = sum_c ctx[r][c]*Wo[o][c] + bo[o]
// ---------------------------------------------------------------------------
__global__ void out_gemm_kernel(const __bf16* __restrict__ ctxb,
                                const __bf16* __restrict__ Wob,
                                const float* __restrict__ bo,
                                float* __restrict__ out) {
  const int lane = threadIdx.x & 31;
  const int w    = blockIdx.x * (blockDim.x >> 5) + (threadIdx.x >> 5);
  const int r0   = (w >> 4) << 4;
  const int o0   = (w & 15) << 4;

  v8f acc = {0, 0, 0, 0, 0, 0, 0, 0};
  for (int k0 = 0; k0 < DMODEL; k0 += 32) {
    v16bf a = ld_tile(ctxb, DMODEL, r0, k0, lane);
    v16bf b = ld_tile(Wob,  DMODEL, o0, k0, lane);
    acc = wmma_bf16(a, b, acc);
  }
  const int rowoff = (lane & 16) >> 1;
  const int col    = lane & 15;
  float bv = bo[o0 + col];
  for (int g = 0; g < 8; ++g)
    out[(size_t)(r0 + g + rowoff) * DMODEL + o0 + col] = acc[g] + bv;
}

// ---------------------------------------------------------------------------
// Launcher
// ---------------------------------------------------------------------------
extern "C" void kernel_launch(void* const* d_in, const int* in_sizes, int n_in,
                              void* d_out, int out_size, void* d_ws, size_t ws_size,
                              hipStream_t stream) {
  const float* s     = (const float*)d_in[0];
  const float* rel_p = (const float*)d_in[1];
  const float* Wq    = (const float*)d_in[2];
  const float* Wk    = (const float*)d_in[3];
  const float* Wv    = (const float*)d_in[4];
  const float* Wm    = (const float*)d_in[5];
  const float* Wo    = (const float*)d_in[6];
  const float* bo    = (const float*)d_in[7];
  float* out = (float*)d_out;

  // Workspace carve-up (all offsets 256B-aligned), total ~19.9 MB
  char* ws = (char*)d_ws;
  __bf16* sb   = (__bf16*)ws;  ws += (size_t)ROWS * DMODEL * 2;        // 512 KB
  __bf16* Wqb  = (__bf16*)ws;  ws += (size_t)DMODEL * DMODEL * 2;      // 128 KB
  __bf16* Wkb  = (__bf16*)ws;  ws += (size_t)DMODEL * DMODEL * 2;
  __bf16* Wvb  = (__bf16*)ws;  ws += (size_t)DMODEL * DMODEL * 2;
  __bf16* Wob  = (__bf16*)ws;  ws += (size_t)DMODEL * DMODEL * 2;
  __bf16* Wmb  = (__bf16*)ws;  ws += (size_t)16 * DMODEL * 2;         // 8 KB (padded)
  __bf16* qb   = (__bf16*)ws;  ws += (size_t)ROWS * DMODEL * 2;        // 512 KB
  __bf16* kb   = (__bf16*)ws;  ws += (size_t)ROWS * DMODEL * 2;
  __bf16* vTb  = (__bf16*)ws;  ws += (size_t)ROWS * DMODEL * 2;
  __bf16* ctxb = (__bf16*)ws;  ws += (size_t)ROWS * DMODEL * 2;
  float*  bias = (float*)ws;   // NBATCH*NHEADS*SEQ*SEQ fp32 = 16 MB

  // K0: bf16 conversions
  cvt_bf16_kernel<<<256, 256, 0, stream>>>(s,  sb,  ROWS * DMODEL);
  cvt_bf16_kernel<<<64,  256, 0, stream>>>(Wq, Wqb, DMODEL * DMODEL);
  cvt_bf16_kernel<<<64,  256, 0, stream>>>(Wk, Wkb, DMODEL * DMODEL);
  cvt_bf16_kernel<<<64,  256, 0, stream>>>(Wv, Wvb, DMODEL * DMODEL);
  cvt_bf16_kernel<<<64,  256, 0, stream>>>(Wo, Wob, DMODEL * DMODEL);
  prep_wm_kernel<<<16, 256, 0, stream>>>(Wm, Wmb);

  // K1: bias projection via WMMA (the 536 MB streaming kernel)
  bias_proj_wmma_kernel<<<2048, 256, 0, stream>>>(rel_p, Wmb, bias);

  // K2: QKV projection, 3072 wave-tasks (384 blocks x 8 waves)
  qkv_gemm_kernel<<<384, 256, 0, stream>>>(sb, Wqb, Wkb, Wvb, qb, kb, vTb);

  // K3: fused attention, one block per (i-tile, head, batch)
  attn_kernel<<<dim3(SEQ / 16, NHEADS, NBATCH), 128, 0, stream>>>(
      qb, kb, vTb, bias, ctxb);

  // K4: output projection + bias, 1024 wave-tasks (128 blocks x 8 waves)
  out_gemm_kernel<<<128, 256, 0, stream>>>(ctxb, Wob, bo, out);
}
